// CRF_36026185679404
// MI455X (gfx1250) — matrix operand order repaired and checked by reading, not compiled, not run on hardware
//
#include <hip/hip_runtime.h>

typedef __attribute__((ext_vector_type(16))) _Float16 v16h;
typedef __attribute__((ext_vector_type(8)))  float    v8f;

#define CRF_B 64
#define CRF_T 1024
#define CRF_K 128
#define CRF_START (CRF_K - 2)
#define BN 16          // batches per block = WMMA N
#define FPAD 132       // padded f32 row (bank spread, keeps 16B alignment)
#define PPAD 136       // padded f16 row (bank spread, keeps 16B alignment)

// ---------------------------------------------------------------------------
// Linear-space CRF forward scan.
// State: u[n][k] = exp(alpha_t[k] - M_t[n]) in f16 LDS (= WMMA B operand),
//        M_t[n] scalar per batch.
// Step:  S[n][j] = sum_k E[j][k] * u[n][k]            (WMMA f16 -> f32)
//        c[n]    = max_j(feat+mt) + log(max_j S)      (1 log/lane)
//        u'[n][j]= exp(feat[j]+mt[j]-c[n]) * S[n][j]  (8 exps/lane)
//        M      += c
// Terminal (t == len-1): fwd = M[n] + log(sum_k u[n][k]).
// A operand (E) is loop-invariant and held in registers across the scan.
// feats[:, t+1, :] staged via global_load_async_to_lds_b128 (ASYNCcnt).
// ---------------------------------------------------------------------------
__global__ __launch_bounds__(256) void crf_forward_kernel(
    const float* __restrict__ feats,
    const int*   __restrict__ lens,
    const float* __restrict__ trans,
    float* __restrict__ fwd_out)
{
  __shared__ _Float16 E_lds[CRF_K][CRF_K];   // exp(trans - rowmax), [j][k]
  __shared__ float    mt_lds[CRF_K];         // rowmax of trans
  __shared__ _Float16 p_lds[BN][PPAD];       // u_t, [n][k]
  __shared__ float    feat_lds[BN][FPAD];    // feats[:, t+1, :]
  __shared__ float    M_lds[BN];             // running log-scale per batch
  __shared__ float    fmax_p[8][BN];         // per-wave partial max of feat+mt
  __shared__ float    smax_p[8][BN];         // per-wave partial max of S

  const int tid  = threadIdx.x;
  const int lane = tid & 31;
  const int w    = tid >> 5;      // wave id 0..7
  const int h    = lane >> 4;     // half-wave 0/1
  const int l16  = lane & 15;
  const int b0   = blockIdx.x * BN;

  // ---- precompute E and mt (threads 0..127 each own one row) ----
  if (tid < CRF_K) {
    const int j = tid;
    float mx = -3.4e38f;
    for (int k = 0; k < CRF_K; ++k) mx = fmaxf(mx, trans[j * CRF_K + k]);
    mt_lds[j] = mx;
    for (int k = 0; k < CRF_K; ++k)
      E_lds[j][k] = (_Float16)__expf(trans[j * CRF_K + k] - mx);
  }
  // ---- init u0 (one-hot at START) and M0 = alpha0[START] ----
  {
    const int n  = tid >> 4;
    const int k0 = (tid & 15) * 8;
    for (int i = 0; i < 8; ++i)
      p_lds[n][k0 + i] = (_Float16)((k0 + i) == CRF_START ? 1.0f : 0.0f);
  }
  if (tid < BN)
    M_lds[tid] = feats[(size_t)(b0 + tid) * CRF_T * CRF_K + CRF_START];
  __syncthreads();

  // ---- hoist loop-invariant A operand (E tiles), mt rows, and len ----
  v16h av[4];
  {
    const int ja = 16 * w + l16;            // A row (M = lane%16)
    #pragma unroll
    for (int kk = 0; kk < 4; ++kk) {
      const int kb = kk * 32;
      #pragma unroll
      for (int i = 0; i < 8; ++i) {         // A 16x32 f16 layout
        av[kk][i]     = E_lds[ja][kb + h * 8 + i];
        av[kk][i + 8] = E_lds[ja][kb + 16 + h * 8 + i];
      }
    }
  }
  float mtv[8];
  #pragma unroll
  for (int r = 0; r < 8; ++r) mtv[r] = mt_lds[16 * w + r + 8 * h];
  const int term_t = (tid < BN) ? (lens[b0 + tid] - 1) : -2;

  for (int t = 0; t < CRF_T; ++t) {
    // ---- A: terminal logsumexp capture (rare, 16 threads, lazy sum of u) ----
    if (term_t == t) {
      float s = 0.f;
      for (int k = 0; k < CRF_K; ++k) s += (float)p_lds[tid][k];
      fwd_out[b0 + tid] = M_lds[tid] + __logf(s);
    }
    if (t == CRF_T - 1) break;

    // ---- stage feats[:, t+1, :] via async DMA to LDS (2 x b128 per thread) ----
    {
      const int n  = tid >> 4;
      const int j0 = (tid & 15) * 8;
      unsigned lds_off = (unsigned)(uintptr_t)(&feat_lds[n][j0]);
      unsigned long long gaddr = (unsigned long long)(uintptr_t)
          (&feats[((size_t)(b0 + n) * CRF_T + (t + 1)) * CRF_K + j0]);
      asm volatile("global_load_async_to_lds_b128 %0, %1, off"
                   :: "v"(lds_off), "v"(gaddr) : "memory");
      asm volatile("global_load_async_to_lds_b128 %0, %1, off offset:16"
                   :: "v"(lds_off), "v"(gaddr) : "memory");
    }

    // ---- B: WMMA GEMM S = E * u ; wave w -> rows j = 16w..16w+15 ----
    v8f acc = {};
    #pragma unroll
    for (int kk = 0; kk < 4; ++kk) {
      const int kb = kk * 32;
      v16h bv;
      #pragma unroll
      for (int i = 0; i < 16; ++i)            // B 32x16 f16 layout (col = lane%16)
        bv[i] = p_lds[l16][kb + h * 8 + i + (i >= 8 ? 8 : 0)];
      acc = __builtin_amdgcn_wmma_f32_16x16x32_f16(
          false, av[kk], false, bv, (short)0, acc, false, false);
    }

    // ---- drain async feats, release p_lds readers ----
    asm volatile("s_wait_asynccnt 0x0" ::: "memory");
    __syncthreads();

    // ---- per-wave partial maxes of (feat+mt) and S for batch n = lane%16 ----
    float valf[8];
    float fm = -3.4e38f, sm = 0.f;
    #pragma unroll
    for (int r = 0; r < 8; ++r) {             // D layout: (M = r + 8h, N = lane%16)
      const int j = 16 * w + r + 8 * h;
      valf[r] = feat_lds[l16][j] + mtv[r];
      fm = fmaxf(fm, valf[r]);
      sm = fmaxf(sm, acc[r]);
    }
    fm = fmaxf(fm, __shfl_xor(fm, 16, 32));   // combine h halves (same n)
    sm = fmaxf(sm, __shfl_xor(sm, 16, 32));
    if (h == 0) { fmax_p[w][l16] = fm; smax_p[w][l16] = sm; }
    __syncthreads();

    // ---- D: reduce partials, rescale, write u_{t+1} ----
    float maxf = -3.4e38f, smax = 0.f;
    #pragma unroll
    for (int ww = 0; ww < 8; ++ww) {
      maxf = fmaxf(maxf, fmax_p[ww][l16]);
      smax = fmaxf(smax, smax_p[ww][l16]);
    }
    smax = fmaxf(smax, 1e-30f);
    const float c = maxf + __logf(smax);      // one log per lane per step
    #pragma unroll
    for (int r = 0; r < 8; ++r) {
      const int j = 16 * w + r + 8 * h;
      p_lds[l16][j] = (_Float16)(__expf(valf[r] - c) * acc[r]);
    }
    if (w == 0 && h == 0) M_lds[l16] += c;    // M_{t+1} = M_t + c
    __syncthreads();
  }
}

// ---------------------------------------------------------------------------
// Gold path score: one wave per batch, lanes stride over T, shuffle-reduce.
// ---------------------------------------------------------------------------
__global__ __launch_bounds__(256) void crf_gold_kernel(
    const float* __restrict__ feats,
    const int*   __restrict__ tags,
    const int*   __restrict__ lens,
    const float* __restrict__ trans,
    float* __restrict__ gold_out)
{
  const int lane = threadIdx.x & 31;
  const int w    = threadIdx.x >> 5;
  const int b    = blockIdx.x * 8 + w;
  if (b >= CRF_B) return;
  const int len = lens[b];
  float s = 0.f;
  for (int i = lane; i < len; i += 32) {
    const int tg = tags[b * CRF_T + i];
    s += feats[((size_t)b * CRF_T + i) * CRF_K + tg];
    if (i + 1 < len) {
      const int tn = tags[b * CRF_T + i + 1];
      s += trans[tn * CRF_K + tg];
    }
  }
  #pragma unroll
  for (int off = 1; off < 32; off <<= 1)
    s += __shfl_xor(s, off, 32);
  if (lane == 0) gold_out[b] = s;
}

// ---------------------------------------------------------------------------
// Final: mean(forward - gold) over B.
// ---------------------------------------------------------------------------
__global__ void crf_final_kernel(const float* __restrict__ fwd,
                                 const float* __restrict__ gold,
                                 float* __restrict__ out)
{
  __shared__ float s[CRF_B];
  const int t = threadIdx.x;
  s[t] = fwd[t] - gold[t];
  __syncthreads();
  for (int off = CRF_B / 2; off > 0; off >>= 1) {
    if (t < off) s[t] += s[t + off];
    __syncthreads();
  }
  if (t == 0) out[0] = s[0] / (float)CRF_B;
}

extern "C" void kernel_launch(void* const* d_in, const int* in_sizes, int n_in,
                              void* d_out, int out_size, void* d_ws, size_t ws_size,
                              hipStream_t stream) {
  (void)in_sizes; (void)n_in; (void)out_size; (void)ws_size;
  const float* feats = (const float*)d_in[0];
  const int*   tags  = (const int*)d_in[1];
  const int*   lens  = (const int*)d_in[2];
  const float* trans = (const float*)d_in[3];
  float* out     = (float*)d_out;
  float* fwd_ws  = (float*)d_ws;          // 64 floats
  float* gold_ws = fwd_ws + CRF_B;        // 64 floats

  crf_forward_kernel<<<CRF_B / BN, 256, 0, stream>>>(feats, lens, trans, fwd_ws);
  crf_gold_kernel<<<CRF_B / 8, 256, 0, stream>>>(feats, tags, lens, trans, gold_ws);
  crf_final_kernel<<<1, CRF_B, 0, stream>>>(fwd_ws, gold_ws, out);
}
